// RegularizedSteinThinning_56487409877231
// MI455X (gfx1250) — compile-verified, matrix operand order checked by reading, not run on hardware
//
#include <hip/hip_runtime.h>

typedef __attribute__((ext_vector_type(16))) _Float16 v16h;
typedef __attribute__((ext_vector_type(8)))  float    v8f;

#define NPTS   500000
#define DIM    32
#define MSEL   1000
#define BLOCKS 256
#define TPB    256
#define NWAVES (TPB / 32)
#define CHUNKS 8                       // 8 chunks of 32 points per wave -> 256 pts/wave
#define PTS_PER_WAVE (CHUNKS * 32)

struct Ws {
    unsigned long long slots[MSEL];    // per-step argmin slot: (orderedKey<<32)|idx
    unsigned           counter;        // monotonic grid barrier counter
    unsigned           pad[63];
};

__global__ void stein_init_ws(Ws* ws) {
    int t = blockIdx.x * blockDim.x + threadIdx.x;
    if (t < MSEL) ws->slots[t] = ~0ull;
    if (t == 0)   ws->counter  = 0u;
}

// one-time f32 -> f16 cache fill (runs only when ws is big enough)
__global__ void stein_cvt(const float* __restrict__ X, const float* __restrict__ S,
                          _Float16* __restrict__ Xh, _Float16* __restrict__ Sh) {
    size_t total = (size_t)NPTS * DIM;
    for (size_t i = (size_t)blockIdx.x * blockDim.x + threadIdx.x; i < total;
         i += (size_t)gridDim.x * blockDim.x) {
        Xh[i] = (_Float16)X[i];
        Sh[i] = (_Float16)S[i];
    }
}

__device__ __forceinline__ unsigned orderKey(float f) {
    unsigned b = __float_as_uint(f);
    return (b & 0x80000000u) ? ~b : (b | 0x80000000u);
}

__device__ __forceinline__ void grid_sync(unsigned* cnt, unsigned target) {
    __syncthreads();
    if (threadIdx.x == 0) {
        __threadfence();
        atomicAdd(cnt, 1u);
        while (atomicAdd(cnt, 0u) < target) __builtin_amdgcn_s_sleep(2);
        __threadfence();
    }
    __syncthreads();
}

__device__ __forceinline__ unsigned long long waveMinKey(unsigned long long k) {
#pragma unroll
    for (int off = 16; off > 0; off >>= 1) {
        unsigned long long o = __shfl_xor(k, off, 32);
        if (o < k) k = o;
    }
    return k;
}

// B fragment (32x16 f16) column loads: lanes 0-15 take K={0..7,16..23},
// lanes 16-31 take K={8..15,24..31}.
__device__ __forceinline__ v16h loadBf(const float* __restrict__ row, int khigh) {
    v16h b;
#pragma unroll
    for (int t = 0; t < 8; ++t) {
        b[t]     = (_Float16)row[khigh + t];
        b[t + 8] = (_Float16)row[16 + khigh + t];
    }
    return b;
}
__device__ __forceinline__ v16h loadBh(const _Float16* __restrict__ row, int khigh) {
    v16h b;
#pragma unroll
    for (int t = 0; t < 8; ++t) {       // two contiguous 16B chunks -> b128 loads
        b[t]     = row[khigh + t];
        b[t + 8] = row[16 + khigh + t];
    }
    return b;
}

template <bool USE_H>
__global__ __launch_bounds__(TPB, 1)
void stein_thin_kernel(const float* __restrict__ X,
                       const float* __restrict__ logp,
                       const float* __restrict__ S,
                       const float* __restrict__ lap,
                       const float* __restrict__ ellp,
                       const _Float16* __restrict__ Xh,
                       const _Float16* __restrict__ Sh,
                       int* __restrict__ out,
                       Ws* __restrict__ ws)
{
    __shared__ float4              stateS[CHUNKS * TPB]; // {nx, xs, wlp, obj} per owned point
    __shared__ float               pxf[DIM], psf[DIM];   // pivot x, s (f32)
    __shared__ _Float16            ph[3 * DIM];          // [x_h | s_h | zeros]
    __shared__ unsigned long long  redW[NWAVES];
    __shared__ unsigned long long  winSh;

    const int tid    = threadIdx.x;
    const int lane   = tid & 31;
    const int waveId = tid >> 5;
    const int r      = lane & 15;
    const int khigh  = (lane & 16) ? 8 : 0;
    const bool hiHalf = (lane >= 16);
    const int gwave     = (blockIdx.x * TPB + tid) >> 5;
    const int waveStart = gwave * PTS_PER_WAVE;

    const float ell      = ellp[0];
    const float ell2     = ell * ell;
    const float inv_ell2 = 1.0f / ell2;
    const float w        = 1.0f / (float)MSEL;

    if (tid < DIM) ph[2 * DIM + tid] = (_Float16)0.0f;   // zero row for A padding

    // ---------------- prologue: per-point state -> LDS ----------------
    unsigned long long lk0 = ~0ull;
    for (int it = 0; it < CHUNKS; ++it) {
        int  j     = waveStart + it * 32 + lane;
        bool valid = (j < NPTS);
        int  jc    = valid ? j : (NPTS - 1);
        const float* xp = X + (size_t)jc * DIM;
        const float* sp = S + (size_t)jc * DIM;
        float a_nx = 0.f, a_ns = 0.f, a_xs = 0.f;
#pragma unroll
        for (int k = 0; k < DIM; ++k) {
            float xv = xp[k], sv = sp[k];
            a_nx += xv * xv; a_ns += sv * sv; a_xs += xv * sv;
        }
        float wl  = w * logp[jc];
        float ob  = valid ? ((float)DIM * inv_ell2 + a_ns + lap[jc] - wl) : 3.0e38f;
        stateS[it * TPB + tid] = make_float4(a_nx, a_xs, wl, ob);
        unsigned long long key = ((unsigned long long)orderKey(ob) << 32) | (unsigned)j;
        lk0 = (key < lk0) ? key : lk0;
    }
    // argmin of obj0 -> slot 0
    {
        unsigned long long wm = waveMinKey(lk0);
        if (lane == 0) redW[waveId] = wm;
        __syncthreads();
        if (tid == 0) {
            unsigned long long bm = redW[0];
            for (int i = 1; i < NWAVES; ++i) { if (redW[i] < bm) bm = redW[i]; }
            atomicMin(&ws->slots[0], bm);
        }
    }
    grid_sync(&ws->counter, (unsigned)BLOCKS);

    // ---------------- m-1 sequential steps ----------------
    for (int s = 1; s < MSEL; ++s) {
        if (tid == 0) winSh = atomicAdd(&ws->slots[s - 1], 0ull);
        __syncthreads();
        const int pidx = (int)(winSh & 0xffffffffu);
        if (blockIdx.x == 0 && tid == 0) out[s - 1] = pidx;

        // stage pivot into LDS (f32 + f16, same RNE rounding as the cache)
        if (tid < DIM) {
            float xv = X[(size_t)pidx * DIM + tid];
            float sv = S[(size_t)pidx * DIM + tid];
            pxf[tid] = xv;           psf[tid] = sv;
            ph[tid]  = (_Float16)xv; ph[DIM + tid] = (_Float16)sv;
        }
        __syncthreads();

        // pivot scalars (redundant per thread, 64 FMA)
        float nxi = 0.f, pxs = 0.f;
#pragma unroll
        for (int k = 0; k < DIM; ++k) { float xv = pxf[k], sv = psf[k]; nxi += xv * xv; pxs += xv * sv; }

        // A fragment: rows {0,8}=x_i, {1,9}=s_i, rest zero
        const int sel = (r == 0 || r == 8) ? 0 : (r == 1 || r == 9) ? 1 : 2;
        const _Float16* ap = ph + sel * DIM;
        v16h A;
#pragma unroll
        for (int t = 0; t < 8; ++t) { A[t] = ap[khigh + t]; A[t + 8] = ap[16 + khigh + t]; }

        const v8f zc = {0.f, 0.f, 0.f, 0.f, 0.f, 0.f, 0.f, 0.f};

        unsigned long long lk = ~0ull;
#pragma unroll 2
        for (int it = 0; it < CHUNKS; ++it) {
            const int cb = waveStart + it * 32;
            int p0 = cb + r;       if (p0 >= NPTS) p0 = NPTS - 1;
            int p1 = cb + 16 + r;  if (p1 >= NPTS) p1 = NPTS - 1;

            v16h bx0, bx1, bs0, bs1;
            if (USE_H) {
                bx0 = loadBh(Xh + (size_t)p0 * DIM, khigh);
                bx1 = loadBh(Xh + (size_t)p1 * DIM, khigh);
                bs0 = loadBh(Sh + (size_t)p0 * DIM, khigh);
                bs1 = loadBh(Sh + (size_t)p1 * DIM, khigh);
            } else {
                bx0 = loadBf(X + (size_t)p0 * DIM, khigh);
                bx1 = loadBf(X + (size_t)p1 * DIM, khigh);
                bs0 = loadBf(S + (size_t)p0 * DIM, khigh);
                bs1 = loadBf(S + (size_t)p1 * DIM, khigh);
            }

            v8f dx0 = __builtin_amdgcn_wmma_f32_16x16x32_f16(false, A, false, bx0, (short)0, zc, false, false);
            v8f dx1 = __builtin_amdgcn_wmma_f32_16x16x32_f16(false, A, false, bx1, (short)0, zc, false, false);
            v8f ds0 = __builtin_amdgcn_wmma_f32_16x16x32_f16(false, A, false, bs0, (short)0, zc, false, false);
            v8f ds1 = __builtin_amdgcn_wmma_f32_16x16x32_f16(false, A, false, bs1, (short)0, zc, false, false);

            // lane L owns point cb+L: low lanes read tile0 rows 0/1, high lanes tile1 rows 8/9
            float xx  = hiHalf ? dx1[0] : dx0[0];   // x_i . x_j
            float sx  = hiHalf ? dx1[1] : dx0[1];   // s_i . x_j
            float xsd = hiHalf ? ds1[0] : ds0[0];   // x_i . s_j
            float ss  = hiHalf ? ds1[1] : ds0[1];   // s_i . s_j

            const int sidx = it * TPB + tid;
            float4 st = stateS[sidx];               // {nx, xs, wlp, obj}

            float r2    = nxi - 2.f * xx + st.x;
            float q     = 1.f + r2 * inv_ell2;
            float rq    = rsqrtf(q);                // q^-0.5
            float qi    = rq * rq;                  // q^-1
            float g     = rq * qi;                  // q^-1.5
            float q25   = g * qi;                   // q^-2.5
            float cross = (pxs - sx - xsd + st.y) * inv_ell2;
            float ki    = (float)DIM * inv_ell2 * g
                        - 3.f * r2 * inv_ell2 * inv_ell2 * q25
                        + cross * g + ss * rq;
            float ob    = st.w + 2.f * ki - st.z;
            stateS[sidx].w = ob;

            unsigned long long key =
                ((unsigned long long)orderKey(ob) << 32) | (unsigned)(cb + lane);
            lk = (key < lk) ? key : lk;
        }

        // wave -> block -> global argmin
        unsigned long long wm = waveMinKey(lk);
        if (lane == 0) redW[waveId] = wm;
        __syncthreads();
        if (tid == 0) {
            unsigned long long bm = redW[0];
            for (int i = 1; i < NWAVES; ++i) { if (redW[i] < bm) bm = redW[i]; }
            atomicMin(&ws->slots[s], bm);
        }
        grid_sync(&ws->counter, (unsigned)((s + 1) * BLOCKS));
    }

    if (blockIdx.x == 0 && tid == 0) {
        unsigned long long win = atomicAdd(&ws->slots[MSEL - 1], 0ull);
        out[MSEL - 1] = (int)(win & 0xffffffffu);
    }
}

extern "C" void kernel_launch(void* const* d_in, const int* in_sizes, int n_in,
                              void* d_out, int out_size, void* d_ws, size_t ws_size,
                              hipStream_t stream) {
    (void)in_sizes; (void)n_in; (void)out_size;
    const float* X    = (const float*)d_in[0];  // x             (N, D)
    const float* logp = (const float*)d_in[1];  // log_p         (N,)
    const float* S    = (const float*)d_in[2];  // score_p       (N, D)
    const float* lap  = (const float*)d_in[3];  // laplace_log_p (N,)
    const float* ell  = (const float*)d_in[4];  // lengthscale   scalar
    int* out = (int*)d_out;
    Ws*  ws  = (Ws*)d_ws;

    const size_t cacheOff   = (sizeof(Ws) + 255) & ~(size_t)255;
    const size_t cacheBytes = (size_t)2 * NPTS * DIM * sizeof(_Float16);  // 64 MB
    const bool   useH       = (ws_size >= cacheOff + cacheBytes);

    _Float16* Xh = (_Float16*)((char*)d_ws + cacheOff);
    _Float16* Sh = Xh + (size_t)NPTS * DIM;

    stein_init_ws<<<dim3(4), dim3(256), 0, stream>>>(ws);
    if (useH) {
        stein_cvt<<<dim3(4096), dim3(256), 0, stream>>>(X, S, Xh, Sh);
        stein_thin_kernel<true><<<dim3(BLOCKS), dim3(TPB), 0, stream>>>(
            X, logp, S, lap, ell, Xh, Sh, out, ws);
    } else {
        stein_thin_kernel<false><<<dim3(BLOCKS), dim3(TPB), 0, stream>>>(
            X, logp, S, lap, ell, Xh, Sh, out, ws);
    }
}
